// VCTRecursiveDecoder_25340307046942
// MI455X (gfx1250) — matrix operand order repaired and checked by reading, not compiled, not run on hardware
//
#include <hip/hip_runtime.h>
#include <cstdint>
#include <cstddef>

// ---------------------------------------------------------------------------
// Types for CDNA5 WMMA (gfx1250, wave32)
// ---------------------------------------------------------------------------
typedef __attribute__((ext_vector_type(16))) __bf16 v16bf;
typedef __attribute__((ext_vector_type(8)))  float  v8f;

union ABfrag { v16bf v; uint4 q[2]; };

__device__ __forceinline__ __bf16 f2bf(float f) {
  union { float f; unsigned u; } x; x.f = f;
  unsigned r = (x.u + 0x7FFFu + ((x.u >> 16) & 1u)) >> 16;   // RNE truncate
  unsigned short h = (unsigned short)r;
  return __builtin_bit_cast(__bf16, h);
}

__device__ __forceinline__ v8f wmma_bf16(const ABfrag& a, const ABfrag& b, v8f c) {
  return __builtin_amdgcn_wmma_f32_16x16x32_bf16(false, a.v, false, b.v,
                                                 (short)0, c, false, false);
}

__device__ __forceinline__ v8f zero8() {
  v8f z = {0.f,0.f,0.f,0.f,0.f,0.f,0.f,0.f};
  return z;
}

// Async global->LDS copy (CDNA5, tracked by ASYNCcnt). lds_off = LDS byte
// address (low 32 bits of generic pointer to __shared__), per lane.
__device__ __forceinline__ void async_ld_b128(unsigned lds_off, const void* gaddr) {
  asm volatile("global_load_async_to_lds_b128 %0, %1, off"
               :: "v"(lds_off), "v"(gaddr) : "memory");
}
__device__ __forceinline__ void wait_async4() {
  asm volatile("s_wait_asynccnt 0x4" ::: "memory");
}
__device__ __forceinline__ void wait_async0() {
  asm volatile("s_wait_asynccnt 0x0" ::: "memory");
}
__device__ __forceinline__ unsigned lds_off(const void* p) {
  return (unsigned)(uintptr_t)p;                             // low 32 bits = LDS addr
}

// ---------------------------------------------------------------------------
// Problem constants
// ---------------------------------------------------------------------------
constexpr int kB      = 16;
constexpr int kC      = 256;
constexpr int kH      = 32;
constexpr int kW      = 32;
constexpr int kBLK    = 16;
constexpr int kL      = 4;
constexpr int kFF     = 1024;
constexpr int kCFEAT  = 64;
constexpr int kCIN    = 3;
constexpr int kGOP    = 4;
constexpr int kPB     = 1024;           // B * NPATCH
constexpr int kTCAP   = 64;             // token capacity (GOP * BLK)
constexpr int kMROWS  = kPB * kTCAP;    // 65536
constexpr int kMEMR   = kPB * kBLK;     // 16384

// ---------------------------------------------------------------------------
// Elementwise helpers
// ---------------------------------------------------------------------------
__global__ void cvt_bf16(const float* __restrict__ s, __bf16* __restrict__ d, int n) {
  int i = blockIdx.x * 256 + threadIdx.x;
  if (i < n) d[i] = f2bf(s[i]);
}

// trg[pb][t][c]: t<16 -> tiled codeword_query, else 0
__global__ void init_trg(const float* __restrict__ query, float* __restrict__ trg) {
  int i = blockIdx.x * 256 + threadIdx.x;                    // over MROWS*C
  int c = i & 255, t = (i >> 8) & 63, pb = i >> 14;
  int p = pb & 63;
  trg[i] = (t < kBLK) ? query[((size_t)p * kBLK + t) * kC + c] : 0.f;
}

// mem[pb][t][c] = codes[b][c][ph*4+wy][pw*4+wx]
__global__ void build_mem(const float* __restrict__ codes, __bf16* __restrict__ mem) {
  int i = blockIdx.x * 256 + threadIdx.x;                    // over MEMR*C
  int c = i & 255, t = (i >> 8) & 15, pb = i >> 12;
  int b = pb >> 6, p = pb & 63, ph = p >> 3, pw = p & 7;
  int wy = t >> 2, wx = t & 3;
  size_t src = (((size_t)b * kC + c) * kH + (ph * 4 + wy)) * kW + (pw * 4 + wx);
  mem[i] = f2bf(codes[src]);
}

__global__ void copy_cvt(const float* __restrict__ s, float* __restrict__ dF,
                         __bf16* __restrict__ dB, int n) {
  int i = blockIdx.x * 256 + threadIdx.x;
  if (i < n) { float v = s[i]; dF[i] = v; dB[i] = f2bf(v); }
}

// trg[:, 16*(i+1)+t] = x[:, 16*i+t]
__global__ void append_y(const float* __restrict__ x, float* __restrict__ trg, int iter) {
  int i = blockIdx.x * 256 + threadIdx.x;                    // over PB*BLK*C
  int c = i & 255, t = (i >> 8) & 15, pb = i >> 12;
  trg[((size_t)pb * kTCAP + kBLK * (iter + 1) + t) * kC + c] =
      x[((size_t)pb * kTCAP + kBLK * iter + t) * kC + c];
}

// yf[b][c][ph*4+wy][pw*4+wx] = x[pb][16*i + wy*4+wx][c]
__global__ void gather_yf(const float* __restrict__ x, float* __restrict__ yf, int iter) {
  int i = blockIdx.x * 256 + threadIdx.x;                    // over B*C*H*W
  int xx = i & 31, yy = (i >> 5) & 31, c = (i >> 10) & 255, b = i >> 18;
  int ph = yy >> 2, wy = yy & 3, pw = xx >> 2, wx = xx & 3;
  int pb = b * 64 + ph * 8 + pw;
  int t  = kBLK * iter + wy * 4 + wx;
  yf[i] = x[((size_t)pb * kTCAP + t) * kC + c];
}

// ---------------------------------------------------------------------------
// bf16 WMMA GEMM:  Y[M][N] = A[M][K] * W[N][K]^T (+bias, opt relu)
// Block 256 threads (8 waves), tile 128x128, K-step 32.
// Double-buffered LDS staged with global_load_async_to_lds_b128 (ASYNCcnt):
// tile k+1 streams into buf^1 while the 8 WMMAs run on buf.
// ---------------------------------------------------------------------------
__global__ __launch_bounds__(256)
void gemm_bf16(const __bf16* __restrict__ A, const __bf16* __restrict__ Wt,
               const float* __restrict__ bias,
               float* __restrict__ outF, __bf16* __restrict__ outB,
               int M, int N, int K, int relu) {
  __shared__ __bf16 lA[2][128][40];   // pitch 80B (16B aligned)
  __shared__ __bf16 lB[2][128][40];

  const int tid  = threadIdx.x;
  const int wave = tid >> 5, lane = tid & 31;
  const int lane16 = lane & 15, half = lane >> 4;
  const int wm = wave & 1, wn = wave >> 1;
  const int m0 = blockIdx.y * 128, n0 = blockIdx.x * 128;

  const int lrow = tid >> 1, lchunk = (tid & 1) * 16;        // 16 bf16 of A + 16 of B per thread

  v8f acc[4][2];
  #pragma unroll
  for (int mt = 0; mt < 4; ++mt)
    #pragma unroll
    for (int nt = 0; nt < 2; ++nt) acc[mt][nt] = zero8();

  const __bf16* gA = A  + (size_t)(m0 + lrow) * K + lchunk;
  const __bf16* gB = Wt + (size_t)(n0 + lrow) * K + lchunk;

  auto issue_tile = [&](int kt, int buf) {
    async_ld_b128(lds_off(&lA[buf][lrow][lchunk]),     gA + kt * 32);
    async_ld_b128(lds_off(&lA[buf][lrow][lchunk + 8]), gA + kt * 32 + 8);
    async_ld_b128(lds_off(&lB[buf][lrow][lchunk]),     gB + kt * 32);
    async_ld_b128(lds_off(&lB[buf][lrow][lchunk + 8]), gB + kt * 32 + 8);
  };

  const int nk = K >> 5;
  issue_tile(0, 0);

  for (int kt = 0; kt < nk; ++kt) {
    const int cur = kt & 1;
    if (kt + 1 < nk) {
      issue_tile(kt + 1, cur ^ 1);
      wait_async4();                 // <=4 left in flight belong to tile kt+1
    } else {
      wait_async0();                 // drain the last tile
    }
    __syncthreads();                 // all waves' chunks of tile kt visible

    ABfrag af[4], bf[2];
    #pragma unroll
    for (int mt = 0; mt < 4; ++mt) {
      int r = wm * 64 + mt * 16 + lane16;
      af[mt].q[0] = *(const uint4*)&lA[cur][r][half * 8];
      af[mt].q[1] = *(const uint4*)&lA[cur][r][half * 8 + 16];
    }
    #pragma unroll
    for (int nt = 0; nt < 2; ++nt) {
      int cidx = wn * 32 + nt * 16 + lane16;
      bf[nt].q[0] = *(const uint4*)&lB[cur][cidx][half * 8];
      bf[nt].q[1] = *(const uint4*)&lB[cur][cidx][half * 8 + 16];
    }
    #pragma unroll
    for (int mt = 0; mt < 4; ++mt)
      #pragma unroll
      for (int nt = 0; nt < 2; ++nt)
        acc[mt][nt] = wmma_bf16(af[mt], bf[nt], acc[mt][nt]);
    __syncthreads();                 // tile kt fully consumed before it is overwritten
  }

  #pragma unroll
  for (int mt = 0; mt < 4; ++mt) {
    #pragma unroll
    for (int nt = 0; nt < 2; ++nt) {
      int mbase = m0 + wm * 64 + mt * 16 + half * 8;
      int n     = n0 + wn * 32 + nt * 16 + lane16;
      float bv  = bias ? bias[n] : 0.f;
      #pragma unroll
      for (int r = 0; r < 8; ++r) {
        float v = acc[mt][nt][r] + bv;
        if (relu) v = fmaxf(v, 0.f);
        size_t o = (size_t)(mbase + r) * N + n;
        if (outF) outF[o] = v;
        if (outB) outB[o] = f2bf(v);
      }
    }
  }
}

// ---------------------------------------------------------------------------
// Self-attention, one wave per (patch-batch, head). T=64, d=32, block-causal
// mask at 16-token (== WMMA tile) granularity.
// qkv: [PB*64][768] bf16 (q|k|v), obuf: [PB*64][256] bf16
// ---------------------------------------------------------------------------
__global__ __launch_bounds__(32)
void attn_self(const __bf16* __restrict__ qkv, __bf16* __restrict__ obuf) {
  __shared__ float  sS[64][72];
  __shared__ __bf16 sP[64][72];
  __shared__ __bf16 sVt[32][72];

  const int lane = threadIdx.x, lane16 = lane & 15, half = lane >> 4;
  const int pb = blockIdx.x >> 3, h = blockIdx.x & 7;
  const __bf16* base = qkv + (size_t)pb * 64 * 768 + h * 32;
  const float scale = 0.17677669529663688f;                  // 1/sqrt(32)

  ABfrag qa[4], kf[4];
  #pragma unroll
  for (int mt = 0; mt < 4; ++mt) {
    const __bf16* row = base + (size_t)(mt * 16 + lane16) * 768;
    qa[mt].q[0] = *(const uint4*)(row + half * 8);
    qa[mt].q[1] = *(const uint4*)(row + half * 8 + 16);
  }
  #pragma unroll
  for (int nt = 0; nt < 4; ++nt) {
    const __bf16* row = base + 256 + (size_t)(nt * 16 + lane16) * 768;
    kf[nt].q[0] = *(const uint4*)(row + half * 8);
    kf[nt].q[1] = *(const uint4*)(row + half * 8 + 16);
  }

  #pragma unroll
  for (int mt = 0; mt < 4; ++mt) {
    #pragma unroll
    for (int nt = 0; nt < 4; ++nt) {
      if (nt <= mt) {                                        // block-causal
        v8f s = wmma_bf16(qa[mt], kf[nt], zero8());
        #pragma unroll
        for (int r = 0; r < 8; ++r)
          sS[mt * 16 + half * 8 + r][nt * 16 + lane16] = s[r] * scale;
      } else {
        #pragma unroll
        for (int r = 0; r < 8; ++r)
          sS[mt * 16 + half * 8 + r][nt * 16 + lane16] = -1e30f;
      }
    }
  }
  __syncthreads();

  #pragma unroll
  for (int rr = 0; rr < 2; ++rr) {                           // 2 rows per lane
    int m = lane + rr * 32;
    float mx = -1e30f;
    for (int j = 0; j < 64; ++j) mx = fmaxf(mx, sS[m][j]);
    float sum = 0.f;
    for (int j = 0; j < 64; ++j) { float e = __expf(sS[m][j] - mx); sS[m][j] = e; sum += e; }
    float inv = 1.f / sum;
    for (int j = 0; j < 64; ++j) sP[m][j] = f2bf(sS[m][j] * inv);
  }
  {                                                          // transpose V
    const __bf16* vb = base + 512 + lane;                    // lane == d index
    for (int t = 0; t < 64; ++t) sVt[lane][t] = vb[(size_t)t * 768];
  }
  __syncthreads();

  #pragma unroll
  for (int mt = 0; mt < 4; ++mt) {
    #pragma unroll
    for (int nt2 = 0; nt2 < 2; ++nt2) {
      v8f o = zero8();
      #pragma unroll
      for (int ks = 0; ks < 2; ++ks) {
        ABfrag pa, vf;
        int row = mt * 16 + lane16;
        pa.q[0] = *(const uint4*)&sP[row][ks * 32 + half * 8];
        pa.q[1] = *(const uint4*)&sP[row][ks * 32 + half * 8 + 16];
        int col = nt2 * 16 + lane16;
        vf.q[0] = *(const uint4*)&sVt[col][ks * 32 + half * 8];
        vf.q[1] = *(const uint4*)&sVt[col][ks * 32 + half * 8 + 16];
        o = wmma_bf16(pa, vf, o);
      }
      #pragma unroll
      for (int r = 0; r < 8; ++r) {
        int m = mt * 16 + half * 8 + r;
        int d = nt2 * 16 + lane16;
        obuf[((size_t)pb * 64 + m) * kC + h * 32 + d] = f2bf(o[r]);
      }
    }
  }
}

// ---------------------------------------------------------------------------
// Cross-attention: queries T=64, keys/values = 16 memory tokens (K padded to 32)
// qc: [PB*64][256], kvc: [PB*16][512] (k|v), obuf: [PB*64][256]
// ---------------------------------------------------------------------------
__global__ __launch_bounds__(32)
void attn_cross(const __bf16* __restrict__ qc, const __bf16* __restrict__ kvc,
                __bf16* __restrict__ obuf) {
  __shared__ float  sS[64][20];
  __shared__ __bf16 sP[64][40];
  __shared__ __bf16 sVt[32][40];

  const int lane = threadIdx.x, lane16 = lane & 15, half = lane >> 4;
  const int pb = blockIdx.x >> 3, h = blockIdx.x & 7;
  const __bf16* qb = qc  + (size_t)pb * 64 * kC + h * 32;
  const __bf16* kb = kvc + (size_t)pb * 16 * 512 + h * 32;        // k: cols [0,256)
  const __bf16* vb = kvc + (size_t)pb * 16 * 512 + 256 + h * 32;  // v: cols [256,512)
  const float scale = 0.17677669529663688f;

  ABfrag qa[4], kf;
  #pragma unroll
  for (int mt = 0; mt < 4; ++mt) {
    const __bf16* row = qb + (size_t)(mt * 16 + lane16) * kC;
    qa[mt].q[0] = *(const uint4*)(row + half * 8);
    qa[mt].q[1] = *(const uint4*)(row + half * 8 + 16);
  }
  {
    const __bf16* row = kb + (size_t)lane16 * 512;
    kf.q[0] = *(const uint4*)(row + half * 8);
    kf.q[1] = *(const uint4*)(row + half * 8 + 16);
  }
  #pragma unroll
  for (int mt = 0; mt < 4; ++mt) {
    v8f s = wmma_bf16(qa[mt], kf, zero8());
    #pragma unroll
    for (int r = 0; r < 8; ++r)
      sS[mt * 16 + half * 8 + r][lane16] = s[r] * scale;
  }
  __syncthreads();

  #pragma unroll
  for (int rr = 0; rr < 2; ++rr) {
    int m = lane + rr * 32;
    float mx = -1e30f;
    for (int j = 0; j < 16; ++j) mx = fmaxf(mx, sS[m][j]);
    float sum = 0.f;
    for (int j = 0; j < 16; ++j) { float e = __expf(sS[m][j] - mx); sS[m][j] = e; sum += e; }
    float inv = 1.f / sum;
    for (int j = 0; j < 16; ++j)  sP[m][j] = f2bf(sS[m][j] * inv);
    for (int j = 16; j < 32; ++j) sP[m][j] = f2bf(0.f);      // K padding
  }
  {
    for (int t = 0; t < 16; ++t)  sVt[lane][t] = vb[(size_t)t * 512 + lane];
    for (int t = 16; t < 32; ++t) sVt[lane][t] = f2bf(0.f);
  }
  __syncthreads();

  #pragma unroll
  for (int mt = 0; mt < 4; ++mt) {
    #pragma unroll
    for (int nt2 = 0; nt2 < 2; ++nt2) {
      ABfrag pa, vf;
      int row = mt * 16 + lane16;
      pa.q[0] = *(const uint4*)&sP[row][half * 8];
      pa.q[1] = *(const uint4*)&sP[row][half * 8 + 16];
      int col = nt2 * 16 + lane16;
      vf.q[0] = *(const uint4*)&sVt[col][half * 8];
      vf.q[1] = *(const uint4*)&sVt[col][half * 8 + 16];
      v8f o = wmma_bf16(pa, vf, zero8());
      #pragma unroll
      for (int r = 0; r < 8; ++r) {
        int m = mt * 16 + half * 8 + r;
        int d = nt2 * 16 + lane16;
        obuf[((size_t)pb * 64 + m) * kC + h * 32 + d] = f2bf(o[r]);
      }
    }
  }
}

// ---------------------------------------------------------------------------
// Fused residual-add + LayerNorm (one wave32 per 256-wide row)
// ---------------------------------------------------------------------------
__global__ __launch_bounds__(256)
void add_ln(const float* __restrict__ a, const float* __restrict__ res,
            const float* __restrict__ w, const float* __restrict__ b,
            float* __restrict__ xout, __bf16* __restrict__ xbf, int rows) {
  int wv = threadIdx.x >> 5, lane = threadIdx.x & 31;
  int row = blockIdx.x * 8 + wv;
  if (row >= rows) return;
  const float* pa = a   + (size_t)row * kC;
  const float* pr = res + (size_t)row * kC;
  float v[8], s = 0.f;
  #pragma unroll
  for (int i = 0; i < 8; ++i) { v[i] = pa[lane + i * 32] + pr[lane + i * 32]; s += v[i]; }
  #pragma unroll
  for (int off = 16; off > 0; off >>= 1) s += __shfl_xor(s, off, 32);
  float mean = s * (1.f / 256.f);
  float vs = 0.f;
  #pragma unroll
  for (int i = 0; i < 8; ++i) { float d = v[i] - mean; vs += d * d; }
  #pragma unroll
  for (int off = 16; off > 0; off >>= 1) vs += __shfl_xor(vs, off, 32);
  float inv = rsqrtf(vs * (1.f / 256.f) + 1e-5f);
  #pragma unroll
  for (int i = 0; i < 8; ++i) {
    int c = lane + i * 32;
    float o = (v[i] - mean) * inv * w[c] + b[c];
    xout[(size_t)row * kC + c] = o;
    xbf [(size_t)row * kC + c] = f2bf(o);
  }
}

// ---------------------------------------------------------------------------
// conv_transpose (stride 2, k=4, SAME pad=1) + relu : (16,256,32,32)->(16,64,64,64)
// ---------------------------------------------------------------------------
__global__ void deconv1_relu(const float* __restrict__ in, const float* __restrict__ w,
                             const float* __restrict__ bias, float* __restrict__ out) {
  size_t idx = (size_t)blockIdx.x * 256 + threadIdx.x;       // 16*64*64*64
  int ox = idx & 63, oy = (idx >> 6) & 63, co = (idx >> 12) & 63;
  int bq = (int)(idx >> 18);
  if (bq >= kB) return;
  float acc = bias[co];
  for (int ky = 0; ky < 4; ++ky) {
    int ty = oy + 1 - ky; if (ty < 0 || (ty & 1)) continue;
    int iy = ty >> 1;     if (iy >= kH) continue;
    for (int kx = 0; kx < 4; ++kx) {
      int tx = ox + 1 - kx; if (tx < 0 || (tx & 1)) continue;
      int ix = tx >> 1;     if (ix >= kW) continue;
      const float* ip = in + (size_t)bq * kC * 1024 + iy * 32 + ix;
      const float* wp = w + co * 16 + ky * 4 + kx;           // w[ci][co][ky][kx]
      float s = 0.f;
      for (int ci = 0; ci < kC; ++ci) s += ip[(size_t)ci * 1024] * wp[(size_t)ci * 1024];
      acc += s;
    }
  }
  out[idx] = fmaxf(acc, 0.f);
}

// (16,64,64,64) -> (16,3,128,128), + sigmoid, written to d_out slice
__global__ void deconv2_sig(const float* __restrict__ in, const float* __restrict__ w,
                            const float* __restrict__ bias, float* __restrict__ out) {
  size_t idx = (size_t)blockIdx.x * 256 + threadIdx.x;       // 16*3*128*128
  int ox = idx & 127, oy = (idx >> 7) & 127;
  int rem = (int)(idx >> 14);
  int co = rem % 3, bq = rem / 3;
  if (bq >= kB) return;
  float acc = bias[co];
  for (int ky = 0; ky < 4; ++ky) {
    int ty = oy + 1 - ky; if (ty < 0 || (ty & 1)) continue;
    int iy = ty >> 1;     if (iy >= 64) continue;
    for (int kx = 0; kx < 4; ++kx) {
      int tx = ox + 1 - kx; if (tx < 0 || (tx & 1)) continue;
      int ix = tx >> 1;     if (ix >= 64) continue;
      const float* ip = in + (size_t)bq * kCFEAT * 4096 + iy * 64 + ix;
      const float* wp = w + co * 16 + ky * 4 + kx;           // w[ci][co][ky][kx]
      float s = 0.f;
      for (int ci = 0; ci < kCFEAT; ++ci) s += ip[(size_t)ci * 4096] * wp[(size_t)ci * 48];
      acc += s;
    }
  }
  out[idx] = 1.f / (1.f + __expf(-acc));
}

// ---------------------------------------------------------------------------
// Host orchestration
// ---------------------------------------------------------------------------
extern "C" void kernel_launch(void* const* d_in, const int* in_sizes, int n_in,
                              void* d_out, int out_size, void* d_ws, size_t ws_size,
                              hipStream_t stream) {
  (void)in_sizes; (void)n_in; (void)out_size; (void)ws_size;
  const float* codes    = (const float*)d_in[0];
  const float* query    = (const float*)d_in[1];
  const float* sa_in_w  = (const float*)d_in[2];
  const float* sa_in_b  = (const float*)d_in[3];
  const float* sa_out_w = (const float*)d_in[4];
  const float* sa_out_b = (const float*)d_in[5];
  const float* ca_in_w  = (const float*)d_in[6];
  const float* ca_in_b  = (const float*)d_in[7];
  const float* ca_out_w = (const float*)d_in[8];
  const float* ca_out_b = (const float*)d_in[9];
  const float* ff_w1    = (const float*)d_in[10];
  const float* ff_b1    = (const float*)d_in[11];
  const float* ff_w2    = (const float*)d_in[12];
  const float* ff_b2    = (const float*)d_in[13];
  const float* ln1_w    = (const float*)d_in[14];
  const float* ln1_b    = (const float*)d_in[15];
  const float* ln2_w    = (const float*)d_in[16];
  const float* ln2_b    = (const float*)d_in[17];
  const float* ln3_w    = (const float*)d_in[18];
  const float* ln3_b    = (const float*)d_in[19];
  const float* dec_w1   = (const float*)d_in[20];
  const float* dec_b1   = (const float*)d_in[21];
  const float* dec_w2   = (const float*)d_in[22];
  const float* dec_b2   = (const float*)d_in[23];
  float* out = (float*)d_out;

  uintptr_t p = (uintptr_t)d_ws;
  auto carve = [&](size_t bytes) -> void* {
    uintptr_t r = (p + 255) & ~(uintptr_t)255; p = r + bytes; return (void*)r;
  };
  __bf16* w_sa_in  = (__bf16*)carve((size_t)kL * 768 * 256 * 2);
  __bf16* w_sa_out = (__bf16*)carve((size_t)kL * 256 * 256 * 2);
  __bf16* w_ca_in  = (__bf16*)carve((size_t)kL * 768 * 256 * 2);
  __bf16* w_ca_out = (__bf16*)carve((size_t)kL * 256 * 256 * 2);
  __bf16* w_ff1    = (__bf16*)carve((size_t)kL * kFF * 256 * 2);
  __bf16* w_ff2    = (__bf16*)carve((size_t)kL * 256 * kFF * 2);
  __bf16* mem_bf   = (__bf16*)carve((size_t)kMEMR * kC * 2);
  float*  trg      = (float*) carve((size_t)kMROWS * kC * 4);
  float*  x        = (float*) carve((size_t)kMROWS * kC * 4);
  __bf16* x_bf     = (__bf16*)carve((size_t)kMROWS * kC * 2);
  __bf16* big      = (__bf16*)carve((size_t)kMROWS * kFF * 2);   // qkv / qc+kvc / hbuf
  __bf16* obuf     = (__bf16*)carve((size_t)kMROWS * kC * 2);
  float*  t0       = (float*) carve((size_t)kMROWS * kC * 4);
  float*  yf       = (float*) carve((size_t)kB * kC * kH * kW * 4);
  float*  h1       = (float*) carve((size_t)kB * kCFEAT * 64 * 64 * 4);

  __bf16* qkv  = big;
  __bf16* qc   = big;
  __bf16* kvc  = big + (size_t)kMROWS * kC;
  __bf16* hbuf = big;

  auto cvt = [&](const float* s, __bf16* d, int n) {
    cvt_bf16<<<(n + 255) / 256, 256, 0, stream>>>(s, d, n);
  };
  auto gemm = [&](const __bf16* A, const __bf16* Wt, const float* bias,
                  float* oF, __bf16* oB, int M, int N, int K, int relu) {
    gemm_bf16<<<dim3(N / 128, M / 128), 256, 0, stream>>>(A, Wt, bias, oF, oB, M, N, K, relu);
  };

  // --- one-time (per launch) conversions -------------------------------
  cvt(sa_in_w,  w_sa_in,  kL * 768 * 256);
  cvt(sa_out_w, w_sa_out, kL * 256 * 256);
  cvt(ca_in_w,  w_ca_in,  kL * 768 * 256);
  cvt(ca_out_w, w_ca_out, kL * 256 * 256);
  cvt(ff_w1,    w_ff1,    kL * kFF * 256);
  cvt(ff_w2,    w_ff2,    kL * 256 * kFF);
  build_mem<<<(kMEMR * kC) / 256, 256, 0, stream>>>(codes, mem_bf);
  init_trg<<<(kMROWS * kC) / 256, 256, 0, stream>>>(query, trg);

  const int nTok = kMROWS * kC;
  for (int i = 0; i < kGOP; ++i) {
    copy_cvt<<<nTok / 256, 256, 0, stream>>>(trg, x, x_bf, nTok);

    for (int l = 0; l < kL; ++l) {
      // ---- self attention ----
      gemm(x_bf, w_sa_in + (size_t)l * 768 * 256, sa_in_b + l * 768,
           nullptr, qkv, kMROWS, 768, 256, 0);
      attn_self<<<kPB * 8, 32, 0, stream>>>(qkv, obuf);
      gemm(obuf, w_sa_out + (size_t)l * 256 * 256, sa_out_b + l * 256,
           t0, nullptr, kMROWS, 256, 256, 0);
      add_ln<<<kMROWS / 8, 256, 0, stream>>>(t0, x, ln1_w + l * 256, ln1_b + l * 256,
                                             x, x_bf, kMROWS);
      // ---- cross attention ----
      gemm(x_bf, w_ca_in + (size_t)l * 768 * 256, ca_in_b + l * 768,
           nullptr, qc, kMROWS, 256, 256, 0);
      gemm(mem_bf, w_ca_in + (size_t)l * 768 * 256 + 256 * 256, ca_in_b + l * 768 + 256,
           nullptr, kvc, kMEMR, 512, 256, 0);
      attn_cross<<<kPB * 8, 32, 0, stream>>>(qc, kvc, obuf);
      gemm(obuf, w_ca_out + (size_t)l * 256 * 256, ca_out_b + l * 256,
           t0, nullptr, kMROWS, 256, 256, 0);
      add_ln<<<kMROWS / 8, 256, 0, stream>>>(t0, x, ln2_w + l * 256, ln2_b + l * 256,
                                             x, x_bf, kMROWS);
      // ---- feed-forward ----
      gemm(x_bf, w_ff1 + (size_t)l * kFF * 256, ff_b1 + l * kFF,
           nullptr, hbuf, kMROWS, kFF, 256, 1);
      gemm(hbuf, w_ff2 + (size_t)l * 256 * kFF, ff_b2 + l * 256,
           t0, nullptr, kMROWS, 256, kFF, 0);
      add_ln<<<kMROWS / 8, 256, 0, stream>>>(t0, x, ln3_w + l * 256, ln3_b + l * 256,
                                             x, x_bf, kMROWS);
    }

    if (i + 1 < kGOP)
      append_y<<<(kPB * kBLK * kC) / 256, 256, 0, stream>>>(x, trg, i);

    // ---- frame decode ----
    gather_yf<<<(kB * kC * kH * kW) / 256, 256, 0, stream>>>(x, yf, i);
    deconv1_relu<<<(kB * kCFEAT * 64 * 64) / 256, 256, 0, stream>>>(yf, dec_w1, dec_b1, h1);
    deconv2_sig<<<(kB * kCIN * 128 * 128) / 256, 256, 0, stream>>>(
        h1, dec_w2, dec_b2, out + (size_t)i * kB * kCIN * 128 * 128);
  }
}
